// GraphAttentionLayer_86053964742796
// MI455X (gfx1250) — compile-verified, matrix operand order checked by reading, not run on hardware
//
#include <hip/hip_runtime.h>

// GAT layer on gfx1250:
//   K1 s1/s2 projections -> K2 column softmax stats -> K3 alpha (f16, normalized)
//   -> K3b X -> f16 -> K4 batched GEMM H = X * alpha^T via v_wmma_f32_16x16x32_f16
//   -> sigmoid.
// Roofline: GEMM = 34.4 GFLOP (f16 WMMA ~15us), min HBM ~300MB (~13us); alpha
// (64MB) + Xh (32MB) live in the 192MB L2. With both operands pre-packed f16,
// BOTH GEMM tiles are pure async global->LDS DMA (ASYNCcnt): the inner loop is
// just ds_load fragments + 8 WMMAs + s_wait_asynccnt/barrier.
// LDS tiles padded to 40 halves/row (80B): 16B chunks stay aligned and the 16
// fragment rows map to disjoint bank quads (r*20 mod 64 all distinct).

typedef __attribute__((ext_vector_type(16))) _Float16 v16h;
typedef __attribute__((ext_vector_type(8)))  _Float16 v8h;
typedef __attribute__((ext_vector_type(4)))  _Float16 v4h;
typedef __attribute__((ext_vector_type(8)))  float    v8f;

#define B_ 32
#define L_ 512
#define N_ 1024
#define LDT 40            // padded LDS tile row stride in halves (80 bytes)
#define NEG_SLOPE 0.01f

// ---------------- Kernel 1: s1[b,n], s2[b,n] -------------------------------
__global__ __launch_bounds__(256)
void gat_sums(const float* __restrict__ X, const float* __restrict__ W,
              float* __restrict__ s1, float* __restrict__ s2) {
  __shared__ float w1s[L_], w2s[L_];
  const int tid = threadIdx.x;
  for (int l = tid; l < L_; l += 256) { w1s[l] = W[l]; w2s[l] = W[L_ + l]; }
  __syncthreads();
  const int n = blockIdx.x * 256 + tid;
  const int b = blockIdx.y;
  const float* xp = X + (size_t)b * L_ * N_ + n;
  float a1 = 0.f, a2 = 0.f;
  for (int l = 0; l < L_; ++l) {
    float x = xp[(size_t)l * N_];
    a1 = fmaf(x, w1s[l], a1);
    a2 = fmaf(x, w2s[l], a2);
  }
  s1[b * N_ + n] = a1;
  s2[b * N_ + n] = a2;
}

// ---------------- Kernel 2: per-(b,j) softmax stats over i -----------------
__global__ __launch_bounds__(256)
void gat_stats(const float* __restrict__ s1, const float* __restrict__ s2,
               float* __restrict__ colmax, float* __restrict__ invden) {
  const int j = blockIdx.x * 256 + threadIdx.x;
  const int b = blockIdx.y;
  const float s2v = s2[b * N_ + j];
  const float* s1p = s1 + b * N_;
  float m = -1e30f;
  for (int i = 0; i < N_; ++i) {
    float e = s1p[i] + s2v;
    e = (e >= 0.f) ? e : NEG_SLOPE * e;
    if (i == j) e = 0.f;
    m = fmaxf(m, e);
  }
  float s = 0.f;
  for (int i = 0; i < N_; ++i) {
    float e = s1p[i] + s2v;
    e = (e >= 0.f) ? e : NEG_SLOPE * e;
    if (i == j) e = 0.f;
    s += __expf(e - m);
  }
  colmax[b * N_ + j] = m;
  invden[b * N_ + j] = 1.f / s;
}

// ---------------- Kernel 3: alpha[b,i,j] (normalized, f16) -----------------
__global__ __launch_bounds__(256)
void gat_alpha(const float* __restrict__ s1, const float* __restrict__ s2,
               const float* __restrict__ colmax, const float* __restrict__ invden,
               _Float16* __restrict__ alpha) {
  const int b = blockIdx.y, i = blockIdx.x;
  const float s1v = s1[b * N_ + i];
  const int j0 = threadIdx.x << 2;
  const float4 s2v = *(const float4*)(s2     + b * N_ + j0);
  const float4 mv  = *(const float4*)(colmax + b * N_ + j0);
  const float4 dv  = *(const float4*)(invden + b * N_ + j0);
  const float* s2a = (const float*)&s2v;
  const float* ma  = (const float*)&mv;
  const float* da  = (const float*)&dv;
  v4h h;
#pragma unroll
  for (int t = 0; t < 4; ++t) {
    float e = s1v + s2a[t];
    e = (e >= 0.f) ? e : NEG_SLOPE * e;
    if (i == j0 + t) e = 0.f;
    h[t] = (_Float16)(__expf(e - ma[t]) * da[t]);
  }
  *(v4h*)(alpha + ((size_t)b * N_ + i) * N_ + j0) = h;
}

// ---------------- Kernel 3b: X (f32) -> Xh (f16), one pass -----------------
__global__ __launch_bounds__(256)
void gat_half(const float* __restrict__ X, _Float16* __restrict__ Xh) {
  const size_t e0 = ((size_t)blockIdx.x * 256 + threadIdx.x) * 8;
  const float4 a = *(const float4*)(X + e0);
  const float4 b = *(const float4*)(X + e0 + 4);
  v8h h;
  h[0] = (_Float16)a.x; h[1] = (_Float16)a.y; h[2] = (_Float16)a.z; h[3] = (_Float16)a.w;
  h[4] = (_Float16)b.x; h[5] = (_Float16)b.y; h[6] = (_Float16)b.z; h[7] = (_Float16)b.w;
  *(v8h*)(Xh + e0) = h;
}

// ---------------- shared GEMM helpers --------------------------------------
__device__ __forceinline__ v16h load_frag(const _Float16* base, int row, int kb) {
  // Documented 16-bit A/B layout: lanes 0-15 K0-7 & 16-23; lanes 16-31 K8-15 & 24-31
  v8h lo = *(const v8h*)(base + row * LDT + kb);
  v8h hi = *(const v8h*)(base + row * LDT + kb + 16);
  return __builtin_shufflevector(lo, hi, 0,1,2,3,4,5,6,7,8,9,10,11,12,13,14,15);
}

__device__ __forceinline__ void store_sigmoid(const v8f (&acc)[2][4],
                                              float* __restrict__ ob,
                                              int l0, int i0, int wm, int wn,
                                              int lane) {
#pragma unroll
  for (int tm = 0; tm < 2; ++tm)
#pragma unroll
    for (int tn = 0; tn < 4; ++tn) {
      const int col = i0 + wn * 64 + tn * 16 + (lane & 15);
      const int row = l0 + wm * 32 + tm * 16 + ((lane < 16) ? 0 : 8);
#pragma unroll
      for (int r = 0; r < 8; ++r) {
        const float v = acc[tm][tn][r];
        ob[(size_t)(row + r) * N_ + col] = 1.f / (1.f + __expf(-v));
      }
    }
}

// Async-copy one 128x32-half tile (row stride N_ halves in global) into a
// padded LDS tile, 16B per lane per issue, 2 issues per thread.
__device__ __forceinline__ void async_tile(const _Float16* __restrict__ src,
                                           _Float16* lds, int k0, int tid) {
#pragma unroll
  for (int q = 0; q < 2; ++q) {
    const int idx = q * 256 + tid;                 // 512 x 16B = 8KB payload
    const int br = idx >> 2, bc = (idx & 3) << 3;
    const unsigned loff = (unsigned)(uintptr_t)(lds + br * LDT + bc);
    const unsigned long long g =
        (unsigned long long)(uintptr_t)(src + (size_t)br * N_ + k0 + bc);
    asm volatile("global_load_async_to_lds_b128 %0, %1, off"
                 :: "v"(loff), "v"(g) : "memory");
  }
}

// ---------------- Kernel 4a: GEMM, both operands f16 via async DMA ---------
// 128(M=l) x 128(N'=i) tile per 256-thread block (8 waves, each 32x64 = 2x4
// WMMA tiles). K(j) step 32. LDS double-buffered; A and B tiles are pure
// async global->LDS copies overlapping the WMMAs.
__global__ __launch_bounds__(256)
void gat_gemm_async(const _Float16* __restrict__ Xh,
                    const _Float16* __restrict__ alpha,
                    float* __restrict__ out) {
  __shared__ __align__(16) _Float16 ldsA[2][128 * LDT];
  __shared__ __align__(16) _Float16 ldsB[2][128 * LDT];

  const int tid  = threadIdx.x;
  const int lane = tid & 31;
  const int wave = tid >> 5;
  const int wm   = wave & 3;
  const int wn   = wave >> 2;
  const int b    = blockIdx.z;
  const int l0   = blockIdx.y * 128;
  const int i0   = blockIdx.x * 128;

  const _Float16* Asrc = Xh    + (size_t)b * L_ * N_ + (size_t)l0 * N_;
  const _Float16* Bsrc = alpha + (size_t)b * N_ * N_ + (size_t)i0 * N_;

  v8f acc[2][4] = {};
  async_tile(Asrc, ldsA[0], 0, tid);
  async_tile(Bsrc, ldsB[0], 0, tid);
  asm volatile("s_wait_asynccnt 0x0" ::: "memory");
  __syncthreads();

  const int kb = (lane < 16) ? 0 : 8;
  for (int ks = 0; ks < N_ / 32; ++ks) {
    const int cur = ks & 1;
    const bool more = (ks + 1) < N_ / 32;
    if (more) {
      async_tile(Asrc, ldsA[cur ^ 1], (ks + 1) * 32, tid);
      async_tile(Bsrc, ldsB[cur ^ 1], (ks + 1) * 32, tid);
    }

    v16h af[2], bf[4];
#pragma unroll
    for (int tm = 0; tm < 2; ++tm)
      af[tm] = load_frag(ldsA[cur], wm * 32 + tm * 16 + (lane & 15), kb);
#pragma unroll
    for (int tn = 0; tn < 4; ++tn)
      bf[tn] = load_frag(ldsB[cur], wn * 64 + tn * 16 + (lane & 15), kb);
#pragma unroll
    for (int tm = 0; tm < 2; ++tm)
#pragma unroll
      for (int tn = 0; tn < 4; ++tn)
        acc[tm][tn] = __builtin_amdgcn_wmma_f32_16x16x32_f16(
            false, af[tm], false, bf[tn], (short)0, acc[tm][tn], false, false);

    if (more) {
      asm volatile("s_wait_asynccnt 0x0" ::: "memory");
      __syncthreads();
    }
  }

  store_sigmoid(acc, out + (size_t)b * L_ * N_, l0, i0, wm, wn, lane);
}

// ---------------- Kernel 4b: fallback fused GEMM (alpha on the fly) --------
__global__ __launch_bounds__(256)
void gat_gemm_fused(const float* __restrict__ X, const float* __restrict__ s1,
                    const float* __restrict__ s2, const float* __restrict__ colmax,
                    const float* __restrict__ invden, float* __restrict__ out) {
  __shared__ __align__(16) _Float16 ldsA[2][128 * LDT];
  __shared__ __align__(16) _Float16 ldsB[2][128 * LDT];

  const int tid  = threadIdx.x;
  const int lane = tid & 31;
  const int wave = tid >> 5;
  const int wm   = wave & 3;
  const int wn   = wave >> 2;
  const int b    = blockIdx.z;
  const int l0   = blockIdx.y * 128;
  const int i0   = blockIdx.x * 128;

  const float* Xb  = X      + (size_t)b * L_ * N_;
  const float* s1b = s1     + b * N_;
  const float* s2b = s2     + b * N_;
  const float* mb  = colmax + b * N_;
  const float* db  = invden + b * N_;

  // s1 rows needed by this thread are wave-uniform and k-invariant: hoist.
  float s1r[16];
#pragma unroll
  for (int t = 0; t < 16; ++t) s1r[t] = s1b[i0 + t * 8 + wave];

  const int arow = tid >> 1;
  const int acol = (tid & 1) << 4;

  float4 ar[4];
  _Float16 bh[16];
  auto loadA = [&](int k0) {
    const float* p = Xb + (size_t)(l0 + arow) * N_ + k0 + acol;
#pragma unroll
    for (int q = 0; q < 4; ++q) ar[q] = *(const float4*)(p + q * 4);
  };
  auto storeA = [&](int buf) {
    _Float16* dst = &ldsA[buf][arow * LDT + acol];
#pragma unroll
    for (int q = 0; q < 4; ++q) {
      v4h h;
      h[0] = (_Float16)ar[q].x; h[1] = (_Float16)ar[q].y;
      h[2] = (_Float16)ar[q].z; h[3] = (_Float16)ar[q].w;
      *(v4h*)(dst + q * 4) = h;
    }
  };
  auto computeB = [&](int k0) {
    const int jj = k0 + lane;
    const float s2v = s2b[jj], mv = mb[jj], dv = db[jj];
#pragma unroll
    for (int t = 0; t < 16; ++t) {
      float e = s1r[t] + s2v;
      e = (e >= 0.f) ? e : NEG_SLOPE * e;
      if (i0 + t * 8 + wave == jj) e = 0.f;
      bh[t] = (_Float16)(__expf(e - mv) * dv);
    }
  };
  auto storeB = [&](int buf) {
#pragma unroll
    for (int t = 0; t < 16; ++t) ldsB[buf][(t * 8 + wave) * LDT + lane] = bh[t];
  };

  v8f acc[2][4] = {};
  loadA(0); computeB(0); storeA(0); storeB(0);
  __syncthreads();

  const int kb = (lane < 16) ? 0 : 8;
  for (int ks = 0; ks < N_ / 32; ++ks) {
    const int cur = ks & 1;
    const bool more = (ks + 1) < N_ / 32;
    if (more) { loadA((ks + 1) * 32); computeB((ks + 1) * 32); }

    v16h af[2], bf[4];
#pragma unroll
    for (int tm = 0; tm < 2; ++tm)
      af[tm] = load_frag(ldsA[cur], wm * 32 + tm * 16 + (lane & 15), kb);
#pragma unroll
    for (int tn = 0; tn < 4; ++tn)
      bf[tn] = load_frag(ldsB[cur], wn * 64 + tn * 16 + (lane & 15), kb);
#pragma unroll
    for (int tm = 0; tm < 2; ++tm)
#pragma unroll
      for (int tn = 0; tn < 4; ++tn)
        acc[tm][tn] = __builtin_amdgcn_wmma_f32_16x16x32_f16(
            false, af[tm], false, bf[tn], (short)0, acc[tm][tn], false, false);

    if (more) {
      storeA(cur ^ 1);
      storeB(cur ^ 1);
      __syncthreads();
    }
  }

  store_sigmoid(acc, out + (size_t)b * L_ * N_, l0, i0, wm, wn, lane);
}

// ---------------- Launch ----------------------------------------------------
extern "C" void kernel_launch(void* const* d_in, const int* in_sizes, int n_in,
                              void* d_out, int out_size, void* d_ws, size_t ws_size,
                              hipStream_t stream) {
  const float* X = (const float*)d_in[0];
  const float* W = (const float*)d_in[1];
  float* out = (float*)d_out;
  float* ws  = (float*)d_ws;
  float* s1  = ws;
  float* s2  = ws + 1 * B_ * N_;
  float* mm  = ws + 2 * B_ * N_;
  float* id  = ws + 3 * B_ * N_;
  const size_t statsBytes = (size_t)4 * B_ * N_ * sizeof(float);
  const size_t alphaBytes = (size_t)B_ * N_ * N_ * sizeof(_Float16);      // 64 MB
  const size_t xhBytes    = (size_t)B_ * L_ * N_ * sizeof(_Float16);      // 32 MB
  _Float16* alpha = (_Float16*)((char*)d_ws + statsBytes);
  _Float16* Xh    = (_Float16*)((char*)d_ws + statsBytes + alphaBytes);

  gat_sums <<<dim3(N_ / 256, B_), 256, 0, stream>>>(X, W, s1, s2);
  gat_stats<<<dim3(N_ / 256, B_), 256, 0, stream>>>(s1, s2, mm, id);

  if (ws_size >= statsBytes + alphaBytes + xhBytes) {
    gat_half <<<dim3((B_ * L_ * N_) / 2048), 256, 0, stream>>>(X, Xh);
    gat_alpha<<<dim3(N_, B_), 256, 0, stream>>>(s1, s2, mm, id, alpha);
    gat_gemm_async<<<dim3(N_ / 128, L_ / 128, B_), 256, 0, stream>>>(Xh, alpha, out);
  } else {
    gat_gemm_fused<<<dim3(N_ / 128, L_ / 128, B_), 256, 0, stream>>>(X, s1, s2, mm, id, out);
  }
}
